// SparseMoEVisionModel_88656714924469
// MI455X (gfx1250) — compile-verified, matrix operand order checked by reading, not run on hardware
//
#include <hip/hip_runtime.h>
#include <hip/hip_bf16.h>

// ---------------- model constants ----------------
#define BATCH   64
#define CIN     3
#define IMGSZ   224
#define PATCHSZ 16
#define DIM     384
#define NHEAD   8
#define HSIZE   48
#define HSP     64          // padded head size for WMMA (K multiple of 32)
#define NLAYER  8
#define NEXP    8
#define FDIM    256
#define GRIDP   14
#define TOK     196
#define BT      (BATCH * TOK)   // 12544
#define FF      (4 * DIM)       // 1536
#define PK      (CIN * PATCHSZ * PATCHSZ) // 768

typedef __attribute__((ext_vector_type(16))) __bf16 v16bf;
typedef __attribute__((ext_vector_type(8)))  __bf16 v8bf;
typedef __attribute__((ext_vector_type(8)))  float  v8f;
typedef __attribute__((ext_vector_type(4)))  unsigned int u32x4;
typedef __attribute__((ext_vector_type(8)))  int  i32x8;
typedef __attribute__((ext_vector_type(4)))  int  i32x4;

#if __has_builtin(__builtin_amdgcn_tensor_load_to_lds)
#define USE_TDM 1
#endif

__device__ __forceinline__ __bf16 f32_to_bf16(float f) {
    unsigned u = __builtin_bit_cast(unsigned, f);
    u += 0x7FFFu + ((u >> 16) & 1u);                 // round-to-nearest-even
    unsigned short h = (unsigned short)(u >> 16);
    return __builtin_bit_cast(__bf16, h);
}

// ---------------- conversion / elementwise helpers ----------------
__global__ void cvt_bf16_kernel(const float* __restrict__ src, __bf16* __restrict__ dst, int n) {
    int idx = blockIdx.x * 256 + threadIdx.x;
    if (idx < n) dst[idx] = f32_to_bf16(src[idx]);
}

__global__ void im2col_bf16_kernel(const float* __restrict__ x, __bf16* __restrict__ col) {
    int idx = blockIdx.x * 256 + threadIdx.x;
    if (idx >= BT * PK) return;
    int k = idx % PK;  int m = idx / PK;
    int b = m / TOK;   int t = m % TOK;
    int gh = t / GRIDP, gw = t % GRIDP;
    int c = k / (PATCHSZ * PATCHSZ);
    int p = (k / PATCHSZ) % PATCHSZ;
    int q = k % PATCHSZ;
    col[idx] = f32_to_bf16(
        x[((size_t)(b * CIN + c) * IMGSZ + (gh * PATCHSZ + p)) * IMGSZ + (gw * PATCHSZ + q)]);
}

__global__ void add_pos_kernel(float* __restrict__ h, const float* __restrict__ pos) {
    int idx = blockIdx.x * 256 + threadIdx.x;
    if (idx >= BT * DIM) return;
    int d = idx % DIM;
    int t = (idx / DIM) % TOK;
    h[idx] += pos[(size_t)t * DIM + d];
}

__global__ void add_kernel(float* __restrict__ a, const float* __restrict__ b, int n) {
    int idx = blockIdx.x * 256 + threadIdx.x;
    if (idx < n) a[idx] += b[idx];
}

__global__ void repack_qkv_kernel(const float* __restrict__ tmp, __bf16* __restrict__ dst) {
    int idx = blockIdx.x * 256 + threadIdx.x;
    if (idx >= BATCH * NHEAD * TOK * HSP) return;
    int hs = idx & (HSP - 1);
    int t  = (idx >> 6) % TOK;
    int bh = idx / (HSP * TOK);
    int h  = bh % NHEAD, b = bh / NHEAD;
    float v = (hs < HSIZE)
        ? tmp[(size_t)(b * TOK + t) * DIM + h * HSIZE + hs] : 0.0f;
    dst[idx] = f32_to_bf16(v);
}

__global__ __launch_bounds__(128)
void layernorm_kernel(const float* __restrict__ x, const float* __restrict__ g,
                      const float* __restrict__ b, float* __restrict__ y,
                      __bf16* __restrict__ yb) {
    int m = blockIdx.x;
    const float* xr = x + (size_t)m * DIM;
    float* yr = y + (size_t)m * DIM;
    __shared__ float red[128];
    int tid = threadIdx.x;
    float s = 0.f;
    for (int i = tid; i < DIM; i += 128) s += xr[i];
    red[tid] = s; __syncthreads();
    for (int st = 64; st > 0; st >>= 1) { if (tid < st) red[tid] += red[tid + st]; __syncthreads(); }
    float mean = red[0] / DIM; __syncthreads();
    float v = 0.f;
    for (int i = tid; i < DIM; i += 128) { float d = xr[i] - mean; v += d * d; }
    red[tid] = v; __syncthreads();
    for (int st = 64; st > 0; st >>= 1) { if (tid < st) red[tid] += red[tid + st]; __syncthreads(); }
    float inv = rsqrtf(red[0] / DIM + 1e-5f);
    for (int i = tid; i < DIM; i += 128) {
        float o = (xr[i] - mean) * inv * g[i] + b[i];
        yr[i] = o;
        if (yb) yb[(size_t)m * DIM + i] = f32_to_bf16(o);
    }
}

__global__ void meanpool_bf16_kernel(const float* __restrict__ y, __bf16* __restrict__ pooled) {
    int idx = blockIdx.x * 256 + threadIdx.x;
    if (idx >= BATCH * DIM) return;
    int b = idx / DIM, d = idx % DIM;
    float s = 0.f;
    for (int t = 0; t < TOK; ++t) s += y[(size_t)(b * TOK + t) * DIM + d];
    pooled[idx] = f32_to_bf16(s * (1.0f / TOK));
}

// ---------------- router / noisy top-2 gating ----------------
__device__ __forceinline__ unsigned pcg(unsigned x) {
    x = x * 747796405u + 2891336453u;
    unsigned w = ((x >> ((x >> 28) + 4u)) ^ x) * 277803737u;
    return (w >> 22) ^ w;
}
__device__ __forceinline__ float nrand(unsigned s) {
    unsigned a = pcg(s), b = pcg(s ^ 0x9E3779B9u);
    float u1 = (a & 0x00FFFFFFu) * (1.0f / 16777216.0f) + 1e-7f;
    float u2 = (b & 0x00FFFFFFu) * (1.0f / 16777216.0f);
    return sqrtf(-2.0f * __logf(u1)) * cosf(6.2831853f * u2);
}

__global__ __launch_bounds__(256)
void router_kernel(const float* __restrict__ xn,
                   const float* __restrict__ Wrt, const float* __restrict__ brt,
                   const float* __restrict__ Wno, const float* __restrict__ bno,
                   float* __restrict__ gate, int layer) {
    int m = blockIdx.x;
    int wave = threadIdx.x >> 5, lane = threadIdx.x & 31;
    __shared__ float slog[NEXP], snoi[NEXP];
    const float* xr = xn + (size_t)m * DIM;
    const float* wr = Wrt + (size_t)wave * DIM;
    const float* wn = Wno + (size_t)wave * DIM;
    float a = 0.f, c = 0.f;
    for (int i = lane; i < DIM; i += 32) { float xv = xr[i]; a += xv * wr[i]; c += xv * wn[i]; }
    for (int d = 1; d < 32; d <<= 1) { a += __shfl_xor(a, d, 32); c += __shfl_xor(c, d, 32); }
    if (lane == 0) { slog[wave] = a + brt[wave]; snoi[wave] = c + bno[wave]; }
    __syncthreads();
    if (threadIdx.x == 0) {
        float noisy[NEXP];
        for (int e = 0; e < NEXP; ++e) {
            float nl = snoi[e];
            float sp = (nl > 20.f) ? nl : log1pf(__expf(nl));
            float eps = nrand((unsigned)(((layer * 1315423911) ^ m) * NEXP + e));
            noisy[e] = slog[e] + eps * sp;
        }
        int i1 = 0, i2 = -1; float v1 = -1e30f, v2 = -1e30f;
        for (int e = 0; e < NEXP; ++e) {
            if (noisy[e] > v1) { v2 = v1; i2 = i1; v1 = noisy[e]; i1 = e; }
            else if (noisy[e] > v2) { v2 = noisy[e]; i2 = e; }
        }
        float e1 = 1.0f, e2 = __expf(v2 - v1), s = e1 + e2;
        for (int e = 0; e < NEXP; ++e) gate[(size_t)m * NEXP + e] = 0.f;
        gate[(size_t)m * NEXP + i1] = e1 / s;
        gate[(size_t)m * NEXP + i2] = e2 / s;
    }
}

// ---------------- Tensor Data Mover tile load (2-D bf16 tile -> LDS) ----------------
__device__ __forceinline__ void tdm_tile_load(unsigned lds_off, const __bf16* g,
                                              int stride_elems, int ext_cols, int ext_rows,
                                              int tile_cols, int tile_rows) {
#ifdef USE_TDM
    unsigned long long ga = (unsigned long long)(size_t)g;
    // D# group0: count=1 | lds_addr | global_addr | type=2 (bits 127:126)
    u32x4 g0 = { 1u, lds_off, (unsigned)ga,
                 (unsigned)((ga >> 32) & 0x01FFFFFFu) | 0x80000000u };
    // D# group1: data_size=1 (2B); tensor_dim0/1; tile_dim0/1; dim0 stride
    i32x8 g1;
    g1[0] = 1 << 16;                                                    // data_size = 2 bytes
    g1[1] = (int)(((unsigned)ext_cols & 0xFFFFu) << 16);                // tensor_dim0 lo16
    g1[2] = (int)((((unsigned)ext_cols) >> 16) |
                  (((unsigned)ext_rows & 0xFFFFu) << 16));              // dim0 hi | dim1 lo
    g1[3] = (int)((((unsigned)ext_rows) >> 16) |
                  ((unsigned)tile_cols << 16));                         // dim1 hi | tile_dim0
    g1[4] = tile_rows;                                                  // tile_dim1, tile_dim2=0
    g1[5] = stride_elems;                                               // tensor_dim0_stride
    g1[6] = 0;
    g1[7] = 0;
    i32x4 z = {0, 0, 0, 0};
#if defined(__clang_major__) && (__clang_major__ >= 23)
    i32x8 z8 = {0, 0, 0, 0, 0, 0, 0, 0};
    __builtin_amdgcn_tensor_load_to_lds(g0, g1, z, z, z8, 0);
#else
    __builtin_amdgcn_tensor_load_to_lds(g0, g1, z, z, 0);
#endif
#endif
}

// ---------------- bf16 WMMA GEMM: C[M,N] = act(A[M,K] * B[N,K]^T + bias) ----------------
#define TM 128
#define TN 64
#define TK 32

__global__ __launch_bounds__(256)
void gemm_bf16_wmma(const __bf16* __restrict__ A, int lda,
                    const __bf16* __restrict__ Bw, int ldb,
                    const float* __restrict__ bias,
                    const float* __restrict__ resid,
                    const float* __restrict__ rowscale, int rs_stride,
                    float* __restrict__ C, __bf16* __restrict__ C16, int ldc,
                    int M, int N, int Kd, int accum, int relu) {
    __shared__ __bf16 As[2][TM][TK];
    __shared__ __bf16 Bs[2][TN][TK];
    int tid = threadIdx.x;
    int wave = tid >> 5, lane = tid & 31;
    int m0 = blockIdx.y * TM;
    int n0 = blockIdx.x * TN;
    v8f acc0 = {}, acc1 = {}, acc2 = {}, acc3 = {};
    int ntiles = Kd / TK;

#ifdef USE_TDM
    if (wave == 0) {  // TDM issue is per-wave, EXEC-independent; uniform operands
        tdm_tile_load((unsigned)(size_t)&As[0][0][0], A + (size_t)m0 * lda, lda,
                      Kd, M - m0, TK, TM);
        tdm_tile_load((unsigned)(size_t)&Bs[0][0][0], Bw + (size_t)n0 * ldb, ldb,
                      Kd, N - n0, TK, TN);
    }
#endif

    for (int it = 0; it < ntiles; ++it) {
        int cur = it & 1;
        int k0 = it * TK;
#ifdef USE_TDM
        if (wave == 0) {
            if (it + 1 < ntiles) {   // pipeline: issue next tiles, wait for oldest pair
                int kn = k0 + TK;
                tdm_tile_load((unsigned)(size_t)&As[1 - cur][0][0],
                              A + (size_t)m0 * lda + kn, lda, Kd - kn, M - m0, TK, TM);
                tdm_tile_load((unsigned)(size_t)&Bs[1 - cur][0][0],
                              Bw + (size_t)n0 * ldb + kn, ldb, Kd - kn, N - n0, TK, TN);
                __builtin_amdgcn_s_wait_tensorcnt(2);
            } else {
                __builtin_amdgcn_s_wait_tensorcnt(0);
            }
        }
#else
        // synchronous staging fallback (bf16 b128 copies)
        #pragma unroll
        for (int i2 = 0; i2 < 2; ++i2) {
            int idx = tid + i2 * 256;          // 512 * 16B = A tile
            int r = idx >> 2, seg = idx & 3;
            uint4 d = {0u, 0u, 0u, 0u};
            if (m0 + r < M) d = *(const uint4*)(A + (size_t)(m0 + r) * lda + k0 + seg * 8);
            *(uint4*)&As[cur][r][seg * 8] = d;
        }
        {
            int r = tid >> 2, seg = tid & 3;   // 256 * 16B = B tile
            uint4 d = {0u, 0u, 0u, 0u};
            if (n0 + r < N) d = *(const uint4*)(Bw + (size_t)(n0 + r) * ldb + k0 + seg * 8);
            *(uint4*)&Bs[cur][r][seg * 8] = d;
        }
#endif
        __syncthreads();

        int rowA = wave * 16 + (lane & 15);
        int kb = (lane >> 4) * 8;              // ISA 16-bit A-layout K split
        v16bf af;
        {
            v8bf lo = *(const v8bf*)&As[cur][rowA][kb];
            v8bf hi = *(const v8bf*)&As[cur][rowA][kb + 16];
            #pragma unroll
            for (int i = 0; i < 8; ++i) { af[i] = lo[i]; af[8 + i] = hi[i]; }
        }
        #pragma unroll
        for (int nf = 0; nf < 4; ++nf) {
            int rowB = nf * 16 + (lane & 15);
            v16bf bfv;
            v8bf lo = *(const v8bf*)&Bs[cur][rowB][kb];
            v8bf hi = *(const v8bf*)&Bs[cur][rowB][kb + 16];
            #pragma unroll
            for (int i = 0; i < 8; ++i) { bfv[i] = lo[i]; bfv[8 + i] = hi[i]; }
            if (nf == 0) acc0 = __builtin_amdgcn_wmma_f32_16x16x32_bf16(false, af, false, bfv, (short)0, acc0, false, false);
            if (nf == 1) acc1 = __builtin_amdgcn_wmma_f32_16x16x32_bf16(false, af, false, bfv, (short)0, acc1, false, false);
            if (nf == 2) acc2 = __builtin_amdgcn_wmma_f32_16x16x32_bf16(false, af, false, bfv, (short)0, acc2, false, false);
            if (nf == 3) acc3 = __builtin_amdgcn_wmma_f32_16x16x32_bf16(false, af, false, bfv, (short)0, acc3, false, false);
        }
        __syncthreads();
    }

    // epilogue: bias / relu / gate-scale / residual / accumulate / bf16 store
    int half = lane >> 4, c16 = lane & 15;
    v8f accs[4] = {acc0, acc1, acc2, acc3};
    #pragma unroll
    for (int nf = 0; nf < 4; ++nf) {
        #pragma unroll
        for (int i = 0; i < 8; ++i) {
            int row = m0 + wave * 16 + i + 8 * half;
            int col = n0 + nf * 16 + c16;
            if (row < M && col < N) {
                float v = accs[nf][i];
                if (bias)     v += bias[col];
                if (relu)     v = fmaxf(v, 0.f);
                if (rowscale) v *= rowscale[(size_t)row * rs_stride];
                if (resid)    v += resid[(size_t)row * ldc + col];
                if (C16)        C16[(size_t)row * ldc + col] = f32_to_bf16(v);
                else if (accum) C[(size_t)row * ldc + col] += v;
                else            C[(size_t)row * ldc + col] = v;
            }
        }
    }
}

// ---------------- flash attention (causal), one wave per 16-row query tile ----------------
__global__ __launch_bounds__(32)
void attn_kernel(const __bf16* __restrict__ qb, const __bf16* __restrict__ kb_,
                 const __bf16* __restrict__ vb, __bf16* __restrict__ out) {
    int bh = blockIdx.x;               // b*NHEAD + h
    int m0 = blockIdx.y * 16;
    int lane = threadIdx.x;
    int half = lane >> 4, c16 = lane & 15;
    int kbse = half * 8;
    const float scale = 0.05103103630798287f;  // DIM^-0.5

    const __bf16* Q = qb + (size_t)bh * TOK * HSP;
    const __bf16* Km = kb_ + (size_t)bh * TOK * HSP;
    const __bf16* Vm = vb + (size_t)bh * TOK * HSP;

    int qrow = m0 + c16; if (qrow >= TOK) qrow = TOK - 1;
    v16bf aq0, aq1;
    {
        v8bf l0 = *(const v8bf*)(Q + (size_t)qrow * HSP + kbse);
        v8bf h0 = *(const v8bf*)(Q + (size_t)qrow * HSP + kbse + 16);
        v8bf l1 = *(const v8bf*)(Q + (size_t)qrow * HSP + 32 + kbse);
        v8bf h1 = *(const v8bf*)(Q + (size_t)qrow * HSP + 32 + kbse + 16);
        #pragma unroll
        for (int i = 0; i < 8; ++i) {
            aq0[i] = l0[i]; aq0[8 + i] = h0[i];
            aq1[i] = l1[i]; aq1[8 + i] = h1[i];
        }
    }

    v8f o0 = {}, o1 = {}, o2 = {};
    float mrow[8], lrow[8];
    #pragma unroll
    for (int i = 0; i < 8; ++i) { mrow[i] = -1e30f; lrow[i] = 0.f; }
    __shared__ __bf16 Pt[16][32];

    int s_last = m0 + 16; if (s_last > TOK) s_last = TOK;
    for (int s0 = 0; s0 < s_last; s0 += 32) {
        v8f sc0 = {}, sc1 = {};
        #pragma unroll
        for (int part = 0; part < 2; ++part) {
            int srow = s0 + part * 16 + c16;
            int srowc = (srow < TOK) ? srow : TOK - 1;
            v16bf b0, b1;
            v8bf l0 = *(const v8bf*)(Km + (size_t)srowc * HSP + kbse);
            v8bf h0 = *(const v8bf*)(Km + (size_t)srowc * HSP + kbse + 16);
            v8bf l1 = *(const v8bf*)(Km + (size_t)srowc * HSP + 32 + kbse);
            v8bf h1 = *(const v8bf*)(Km + (size_t)srowc * HSP + 32 + kbse + 16);
            #pragma unroll
            for (int i = 0; i < 8; ++i) {
                b0[i] = l0[i]; b0[8 + i] = h0[i];
                b1[i] = l1[i]; b1[8 + i] = h1[i];
            }
            if (part == 0) {
                sc0 = __builtin_amdgcn_wmma_f32_16x16x32_bf16(false, aq0, false, b0, (short)0, sc0, false, false);
                sc0 = __builtin_amdgcn_wmma_f32_16x16x32_bf16(false, aq1, false, b1, (short)0, sc0, false, false);
            } else {
                sc1 = __builtin_amdgcn_wmma_f32_16x16x32_bf16(false, aq0, false, b0, (short)0, sc1, false, false);
                sc1 = __builtin_amdgcn_wmma_f32_16x16x32_bf16(false, aq1, false, b1, (short)0, sc1, false, false);
            }
        }
        // scale + causal mask + online softmax
        float alpha[8];
        #pragma unroll
        for (int i = 0; i < 8; ++i) {
            int row = m0 + i + 8 * half;
            float v0 = sc0[i] * scale, v1 = sc1[i] * scale;
            int col0 = s0 + c16, col1 = s0 + 16 + c16;
            if (col0 > row || col0 >= TOK) v0 = -1e30f;
            if (col1 > row || col1 >= TOK) v1 = -1e30f;
            float mx = fmaxf(v0, v1);
            for (int d = 1; d < 16; d <<= 1) mx = fmaxf(mx, __shfl_xor(mx, d, 32));
            float mnew = fmaxf(mrow[i], mx);
            alpha[i] = __expf(mrow[i] - mnew);
            float p0 = __expf(v0 - mnew), p1 = __expf(v1 - mnew);
            sc0[i] = p0; sc1[i] = p1;
            float sum = p0 + p1;
            for (int d = 1; d < 16; d <<= 1) sum += __shfl_xor(sum, d, 32);
            lrow[i] = lrow[i] * alpha[i] + sum;
            mrow[i] = mnew;
        }
        #pragma unroll
        for (int i = 0; i < 8; ++i) { o0[i] *= alpha[i]; o1[i] *= alpha[i]; o2[i] *= alpha[i]; }
        // bounce P (16x32) through LDS to rebuild A-fragment layout
        #pragma unroll
        for (int i = 0; i < 8; ++i) {
            int row = i + 8 * half;
            Pt[row][c16]      = f32_to_bf16(sc0[i]);
            Pt[row][16 + c16] = f32_to_bf16(sc1[i]);
        }
        __syncthreads();
        v16bf pa;
        {
            v8bf lo = *(const v8bf*)&Pt[c16][kbse];
            v8bf hi = *(const v8bf*)&Pt[c16][kbse + 16];
            #pragma unroll
            for (int i = 0; i < 8; ++i) { pa[i] = lo[i]; pa[8 + i] = hi[i]; }
        }
        __syncthreads();
        // P(16x32) x V(32x48)
        #pragma unroll
        for (int nf = 0; nf < 3; ++nf) {
            v16bf bv;
            #pragma unroll
            for (int i = 0; i < 8; ++i) {
                int sa = s0 + kbse + i;      if (sa >= TOK) sa = TOK - 1;
                int sb = s0 + kbse + 16 + i; if (sb >= TOK) sb = TOK - 1;
                bv[i]     = Vm[(size_t)sa * HSP + nf * 16 + c16];
                bv[8 + i] = Vm[(size_t)sb * HSP + nf * 16 + c16];
            }
            if (nf == 0) o0 = __builtin_amdgcn_wmma_f32_16x16x32_bf16(false, pa, false, bv, (short)0, o0, false, false);
            if (nf == 1) o1 = __builtin_amdgcn_wmma_f32_16x16x32_bf16(false, pa, false, bv, (short)0, o1, false, false);
            if (nf == 2) o2 = __builtin_amdgcn_wmma_f32_16x16x32_bf16(false, pa, false, bv, (short)0, o2, false, false);
        }
    }

    int b = bh / NHEAD, h = bh % NHEAD;
    #pragma unroll
    for (int i = 0; i < 8; ++i) {
        int row = m0 + i + 8 * half;
        if (row < TOK) {
            float inv = 1.0f / lrow[i];
            size_t base = ((size_t)(b * TOK + row) * NHEAD + h) * HSIZE;
            out[base + 0  + c16] = f32_to_bf16(o0[i] * inv);
            out[base + 16 + c16] = f32_to_bf16(o1[i] * inv);
            out[base + 32 + c16] = f32_to_bf16(o2[i] * inv);
        }
    }
}

// ---------------- host orchestration ----------------
static inline void gemm(hipStream_t st, const __bf16* A, int lda, const __bf16* Bw, int ldb,
                        const float* bias, const float* resid,
                        const float* rowscale, int rs_stride,
                        float* C, __bf16* C16, int ldc, int M, int N, int Kd,
                        int accum, int relu) {
    dim3 g((N + TN - 1) / TN, (M + TM - 1) / TM);
    gemm_bf16_wmma<<<g, 256, 0, st>>>(A, lda, Bw, ldb, bias, resid, rowscale, rs_stride,
                                      C, C16, ldc, M, N, Kd, accum, relu);
}
static inline int blks(long long n) { return (int)((n + 255) / 256); }
static inline void cvt(hipStream_t st, const float* src, __bf16* dst, long long n) {
    cvt_bf16_kernel<<<blks(n), 256, 0, st>>>(src, dst, (int)n);
}

extern "C" void kernel_launch(void* const* d_in, const int* in_sizes, int n_in,
                              void* d_out, int out_size, void* d_ws, size_t ws_size,
                              hipStream_t stream) {
    const float* x      = (const float*)d_in[0];
    const float* Wpatch = (const float*)d_in[1];
    const float* bpatch = (const float*)d_in[2];
    const float* pos    = (const float*)d_in[3];
    const float* Wq     = (const float*)d_in[4];
    const float* Wk     = (const float*)d_in[5];
    const float* Wv     = (const float*)d_in[6];
    const float* Wproj  = (const float*)d_in[7];
    const float* bproj  = (const float*)d_in[8];
    const float* ln1g   = (const float*)d_in[9];
    const float* ln1b   = (const float*)d_in[10];
    const float* ln2g   = (const float*)d_in[11];
    const float* ln2b   = (const float*)d_in[12];
    const float* Wrt    = (const float*)d_in[13];
    const float* brt    = (const float*)d_in[14];
    const float* Wno    = (const float*)d_in[15];
    const float* bno    = (const float*)d_in[16];
    const float* We1    = (const float*)d_in[17];
    const float* be1    = (const float*)d_in[18];
    const float* We2    = (const float*)d_in[19];
    const float* be2    = (const float*)d_in[20];
    const float* lnfg   = (const float*)d_in[21];
    const float* lnfb   = (const float*)d_in[22];
    const float* Whead  = (const float*)d_in[23];
    const float* bhead  = (const float*)d_in[24];

    // workspace layout (256B aligned chunks)
    char* w = (char*)d_ws;
    #define WS_ALLOC(ty, name, elems) \
        ty* name = (ty*)w; w += (((size_t)(elems) * sizeof(ty)) + 255) & ~(size_t)255;
    WS_ALLOC(float,  s_h,    (size_t)BT * DIM)
    WS_ALLOC(float,  s_xn,   (size_t)BT * DIM)
    WS_ALLOC(float,  s_tmp,  (size_t)BT * DIM)
    WS_ALLOC(float,  s_moe,  (size_t)BT * DIM)
    WS_ALLOC(__bf16, s_xnb,  (size_t)BT * DIM)
    WS_ALLOC(__bf16, s_attb, (size_t)BT * DIM)
    WS_ALLOC(__bf16, s_colb, (size_t)BT * PK)
    WS_ALLOC(__bf16, s_hidb, (size_t)BT * FF)
    WS_ALLOC(__bf16, s_q,    (size_t)BATCH * NHEAD * TOK * HSP)
    WS_ALLOC(__bf16, s_k,    (size_t)BATCH * NHEAD * TOK * HSP)
    WS_ALLOC(__bf16, s_v,    (size_t)BATCH * NHEAD * TOK * HSP)
    WS_ALLOC(float,  s_gate, (size_t)BT * NEXP)
    WS_ALLOC(__bf16, s_poolb,(size_t)BATCH * DIM)
    WS_ALLOC(__bf16, wPatch, (size_t)DIM * PK)
    WS_ALLOC(__bf16, wQ,     (size_t)NLAYER * DIM * DIM)
    WS_ALLOC(__bf16, wK,     (size_t)NLAYER * DIM * DIM)
    WS_ALLOC(__bf16, wV,     (size_t)NLAYER * DIM * DIM)
    WS_ALLOC(__bf16, wP,     (size_t)NLAYER * DIM * DIM)
    WS_ALLOC(__bf16, wE1,    (size_t)NLAYER * NEXP * FF * DIM)
    WS_ALLOC(__bf16, wE2,    (size_t)NLAYER * NEXP * DIM * FF)
    WS_ALLOC(__bf16, wHead,  (size_t)FDIM * DIM)
    #undef WS_ALLOC

    // ---- one-time (per launch) weight conversion to bf16 ----
    cvt(stream, Wpatch, wPatch, (long long)DIM * PK);
    cvt(stream, Wq, wQ, (long long)NLAYER * DIM * DIM);
    cvt(stream, Wk, wK, (long long)NLAYER * DIM * DIM);
    cvt(stream, Wv, wV, (long long)NLAYER * DIM * DIM);
    cvt(stream, Wproj, wP, (long long)NLAYER * DIM * DIM);
    cvt(stream, We1, wE1, (long long)NLAYER * NEXP * FF * DIM);
    cvt(stream, We2, wE2, (long long)NLAYER * NEXP * DIM * FF);
    cvt(stream, Whead, wHead, (long long)FDIM * DIM);

    // ---- patch embedding ----
    im2col_bf16_kernel<<<blks((long long)BT * PK), 256, 0, stream>>>(x, s_colb);
    gemm(stream, s_colb, PK, wPatch, PK, bpatch, nullptr, nullptr, 0,
         s_h, nullptr, DIM, BT, DIM, PK, 0, 0);
    add_pos_kernel<<<blks((long long)BT * DIM), 256, 0, stream>>>(s_h, pos);

    const int attn_tiles = (TOK + 15) / 16;
    for (int l = 0; l < NLAYER; ++l) {
        // attention
        layernorm_kernel<<<BT, 128, 0, stream>>>(s_h, ln1g + l * DIM, ln1b + l * DIM, s_xn, s_xnb);
        gemm(stream, s_xnb, DIM, wQ + (size_t)l * DIM * DIM, DIM, nullptr, nullptr, nullptr, 0,
             s_tmp, nullptr, DIM, BT, DIM, DIM, 0, 0);
        repack_qkv_kernel<<<blks((long long)BATCH * NHEAD * TOK * HSP), 256, 0, stream>>>(s_tmp, s_q);
        gemm(stream, s_xnb, DIM, wK + (size_t)l * DIM * DIM, DIM, nullptr, nullptr, nullptr, 0,
             s_tmp, nullptr, DIM, BT, DIM, DIM, 0, 0);
        repack_qkv_kernel<<<blks((long long)BATCH * NHEAD * TOK * HSP), 256, 0, stream>>>(s_tmp, s_k);
        gemm(stream, s_xnb, DIM, wV + (size_t)l * DIM * DIM, DIM, nullptr, nullptr, nullptr, 0,
             s_tmp, nullptr, DIM, BT, DIM, DIM, 0, 0);
        repack_qkv_kernel<<<blks((long long)BATCH * NHEAD * TOK * HSP), 256, 0, stream>>>(s_tmp, s_v);

        attn_kernel<<<dim3(BATCH * NHEAD, attn_tiles), 32, 0, stream>>>(s_q, s_k, s_v, s_attb);

        // h = h + att @ Wproj^T + bproj
        gemm(stream, s_attb, DIM, wP + (size_t)l * DIM * DIM, DIM,
             bproj + l * DIM, s_h, nullptr, 0, s_h, nullptr, DIM, BT, DIM, DIM, 0, 0);

        // MoE
        layernorm_kernel<<<BT, 128, 0, stream>>>(s_h, ln2g + l * DIM, ln2b + l * DIM, s_xn, s_xnb);
        router_kernel<<<BT, 256, 0, stream>>>(s_xn, Wrt + (size_t)l * NEXP * DIM, brt + l * NEXP,
                                              Wno + (size_t)l * NEXP * DIM, bno + l * NEXP,
                                              s_gate, l);
        hipMemsetAsync(s_moe, 0, (size_t)BT * DIM * sizeof(float), stream);
        for (int e = 0; e < NEXP; ++e) {
            const __bf16* w1 = wE1 + ((size_t)l * NEXP + e) * FF * DIM;
            const float*  b1 = be1 + ((size_t)l * NEXP + e) * FF;
            const __bf16* w2 = wE2 + ((size_t)l * NEXP + e) * DIM * FF;
            const float*  b2 = be2 + ((size_t)l * NEXP + e) * DIM;
            gemm(stream, s_xnb, DIM, w1, DIM, b1, nullptr, nullptr, 0,
                 nullptr, s_hidb, FF, BT, FF, DIM, 0, 1);              // relu, bf16 out
            gemm(stream, s_hidb, FF, w2, FF, b2, nullptr, s_gate + e, NEXP,
                 s_moe, nullptr, DIM, BT, DIM, FF, 1, 0);              // gated accumulate
        }
        add_kernel<<<blks((long long)BT * DIM), 256, 0, stream>>>(s_h, s_moe, BT * DIM);
    }

    // final LN + mean pool + head
    layernorm_kernel<<<BT, 128, 0, stream>>>(s_h, lnfg, lnfb, s_xn, s_xnb);
    meanpool_bf16_kernel<<<blks((long long)BATCH * DIM), 256, 0, stream>>>(s_xn, s_poolb);
    gemm(stream, s_poolb, DIM, wHead, DIM, bhead, nullptr, nullptr, 0,
         (float*)d_out, nullptr, FDIM, BATCH, FDIM, DIM, 0, 0);
}
